// EngramLayer_23570780520532
// MI455X (gfx1250) — matrix operand order repaired and checked by reading, not compiled, not run on hardware
//
#include <hip/hip_runtime.h>

// ---------------- problem constants ----------------
#define D       64
#define TLEN    8192
#define BATCH   32
#define NVOCAB  500

// ---------------- tiling ----------------
#define WAVES   6
#define WOUT    48                 // output tokens per wave
#define WG_TOK  (WAVES * WOUT)     // 288
#define ROWS_X  (WG_TOK + 18)      // halo 16 + window lookback 2

// ---------------- workspace (global scratch) layout, bytes ----------------
#define WS_EMBT 0                  // f16 emb table [500][64]  (64000 B)
#define WS_W12  65536              // 64  frags * 1KB
#define WS_W13  131072             // 96  frags * 1KB
#define WS_W22  229376             // 32  frags * 1KB
#define WS_W23  262144             // 32  frags * 1KB
#define WS_KEY  294912             // 16  frags * 1KB
#define WS_VAL  311296             // 16  frags * 1KB
#define WS_TOTAL 327680

// ---------------- LDS layout, bytes ----------------
#define SM_WFRAG 0                 // W1_2 (64KB) then W1_3 (96KB) fragment-major
#define SM_W13   65536
#define SM_XT    163840            // f16 x-tile [ROWS_X][64]  (39168 B)
#define SM_WAVE0 203008
#define PW       20736             // per-wave scratch
#define O_EMB    0                 // f16 emb [16][136]  /  alias: f16 vn [64][64]
#define O_KH     8192              // f32 kbuf [16][66] (4224B) / alias f16 Hs [16][36]
#define O_GATE   (O_KH + 4224)     // f32 gate[16]
#define O_VAL    12544             // f16 value [64][64]
#define SMEM_TOTAL (SM_WAVE0 + WAVES * PW)   // 327424
static_assert(SMEM_TOTAL <= 320 * 1024, "LDS budget");

typedef _Float16 f16;
typedef __attribute__((ext_vector_type(16))) _Float16 v16h;
typedef __attribute__((ext_vector_type(8)))  float    v8f;
typedef __attribute__((ext_vector_type(4)))  unsigned u32x4;
typedef __attribute__((ext_vector_type(8)))  int      i32x8;
typedef __attribute__((ext_vector_type(4)))  int      i32x4;

union FragU { unsigned u[8]; v16h h; };

#if defined(__has_builtin)
#if __has_builtin(__builtin_amdgcn_tensor_load_to_lds)
#define HAVE_TDM 1
#endif
#endif

__device__ __forceinline__ v8f wmma16(v16h a, v16h b, v8f c) {
  return __builtin_amdgcn_wmma_f32_16x16x32_f16(false, a, false, b, (short)0, c,
                                                false, false);
}

// B fragment (ISA 16-bit 32x16 layout) pre-packed fragment-major: lane L owns
// 32 contiguous bytes -> two b128 loads (works for LDS and global/L2 sources).
__device__ __forceinline__ v16h load_bfrag(const char* p, int lane) {
  FragU f;
  const uint4* q = (const uint4*)(p + lane * 32);
  uint4 a = q[0], b = q[1];
  f.u[0] = a.x; f.u[1] = a.y; f.u[2] = a.z; f.u[3] = a.w;
  f.u[4] = b.x; f.u[5] = b.y; f.u[6] = b.z; f.u[7] = b.w;
  return f.h;
}

// A fragment straight out of the x-tile: window matrices are views with row
// stride 64 f16 (win_n[t] = 64*n consecutive f16 starting at token t-(n-1)).
__device__ __forceinline__ v16h load_afrag_win(const f16* xrow, int k0, int lane) {
  FragU f;
  int M  = lane & 15;
  int kb = (lane & 16) ? 8 : 0;                // ISA 16-bit A layout
#pragma unroll
  for (int j = 0; j < 8; ++j) {
    int k = k0 + ((j < 4) ? 2 * j : 16 + 2 * (j - 4)) + kb;
    f.u[j] = *(const unsigned*)(xrow + (M + (k >> 6)) * 64 + (k & 63));
  }
  return f.h;
}

// A fragment from a [16][stride] f16 LDS panel, one K=32 chunk.
__device__ __forceinline__ v16h load_afrag_row(const f16* base, int stride, int lane) {
  FragU f;
  int M  = lane & 15;
  int kb = (lane & 16) ? 8 : 0;
#pragma unroll
  for (int j = 0; j < 8; ++j) {
    int k = ((j < 4) ? 2 * j : 16 + 2 * (j - 4)) + kb;
    f.u[j] = *(const unsigned*)(base + M * stride + k);
  }
  return f.h;
}

__device__ __forceinline__ float siluf(float x) { return x / (1.f + __expf(-x)); }

// ================= prep: f16 emb table + fragment-major f16 weights =========
__global__ __launch_bounds__(256) void engram_prep(
    const float* __restrict__ W12, const float* __restrict__ W13,
    const float* __restrict__ W22, const float* __restrict__ W23,
    const float* __restrict__ Wk,  const float* __restrict__ Wv,
    const float* __restrict__ embf, char* __restrict__ ws)
{
  int blk = blockIdx.x, tid = threadIdx.x;
  if (blk < 256) {                         // one block = one 16x32 B fragment
    const float* W; int KT, N, off, fi;
    if      (blk < 64)  { W = W12; KT = 4; N = 256; off = WS_W12; fi = blk;       }
    else if (blk < 160) { W = W13; KT = 6; N = 256; off = WS_W13; fi = blk - 64;  }
    else if (blk < 192) { W = W22; KT = 8; N = 64;  off = WS_W22; fi = blk - 160; }
    else if (blk < 224) { W = W23; KT = 8; N = 64;  off = WS_W23; fi = blk - 192; }
    else if (blk < 240) { W = Wk;  KT = 4; N = 64;  off = WS_KEY; fi = blk - 224; }
    else                { W = Wv;  KT = 4; N = 64;  off = WS_VAL; fi = blk - 240; }
    int nt = fi / KT, kt = fi - nt * KT;
    int lane = tid >> 3, j = tid & 7;
    int k0 = kt * 32 + ((j < 4) ? 2 * j : 16 + 2 * (j - 4)) + ((lane & 16) ? 8 : 0);
    int n  = nt * 16 + (lane & 15);
    f16* dst = (f16*)(ws + off + fi * 1024 + lane * 32 + j * 4);
    dst[0] = (f16)W[(size_t)k0 * N + n];
    dst[1] = (f16)W[(size_t)(k0 + 1) * N + n];
  } else {                                 // f16 embedding table
    int idx = (blk - 256) * 256 + tid;
    if (idx < (NVOCAB * D) / 2) {
      f16* dst = (f16*)(ws + WS_EMBT + idx * 4);
      dst[0] = (f16)embf[idx * 2];
      dst[1] = (f16)embf[idx * 2 + 1];
    }
  }
}

// ================= fused engram layer ========================================
__global__ __launch_bounds__(WAVES * 32) void engram_kernel(
    const float* __restrict__ hidden, const int* __restrict__ ids,
    const char* __restrict__ ws,
    const float* __restrict__ b12, const float* __restrict__ b22,
    const float* __restrict__ b13, const float* __restrict__ b23,
    const float* __restrict__ keyb, const float* __restrict__ valb,
    const float* __restrict__ n1g, const float* __restrict__ n1b,
    const float* __restrict__ n2g, const float* __restrict__ n2b,
    const float* __restrict__ convw, const float* __restrict__ cng,
    const float* __restrict__ cnb, float* __restrict__ out)
{
  extern __shared__ char smem[];
  const int b    = blockIdx.y;
  const int g0   = blockIdx.x * WG_TOK;
  const int tid  = threadIdx.x;
  const int wv   = tid >> 5;
  const int lane = tid & 31;

  // ---- gather f16 embeddings for all tokens this WG touches (with halo) ----
  {
    const uint4* embT = (const uint4*)(ws + WS_EMBT);
    uint4* xt4 = (uint4*)(smem + SM_XT);
    for (int i = tid; i < ROWS_X; i += WAVES * 32) {
      int tt = g0 - 18 + i;
      if (tt >= 0 && tt < TLEN) {
        const uint4* src = embT + (size_t)ids[(size_t)b * TLEN + tt] * 8;
#pragma unroll
        for (int c = 0; c < 8; ++c) xt4[i * 8 + c] = src[c];
      } else {
        uint4 z; z.x = z.y = z.z = z.w = 0u;
#pragma unroll
        for (int c = 0; c < 8; ++c) xt4[i * 8 + c] = z;
      }
    }
  }

  // ---- stage the two W1 fragment arrays (160KB) into LDS ----
#if HAVE_TDM
  if (wv == 0) {
    unsigned lds_off = (unsigned)(size_t)(smem + SM_WFRAG);   // low 32b = LDS offset
    unsigned long long ga = (unsigned long long)(size_t)(ws + WS_W12);
    u32x4 g0d = { 1u,                                   // count=1, user desc
                  lds_off,
                  (unsigned)ga,
                  (unsigned)((ga >> 32) & 0x1ffffffu) | (2u << 30) };   // type=2
    // 1-D copy: 20480 units of 8 bytes (tile_dim0 = tensor_dim0 = 20480)
    i32x8 g1d = { (int)(3u << 16),           // data_size = 8B
                  (int)(20480u << 16),       // tensor_dim0[15:0]
                  0,
                  (int)(20480u << 16),       // tile_dim0
                  0, 20480, 0, 0 };
    i32x4 z4 = { 0, 0, 0, 0 };
#if defined(__clang_major__) && (__clang_major__ >= 23)
    i32x8 z8 = { 0, 0, 0, 0, 0, 0, 0, 0 };
    __builtin_amdgcn_tensor_load_to_lds(g0d, g1d, z4, z4, z8, 0);
#else
    __builtin_amdgcn_tensor_load_to_lds(g0d, g1d, z4, z4, 0);
#endif
#if __has_builtin(__builtin_amdgcn_s_wait_tensorcnt)
    __builtin_amdgcn_s_wait_tensorcnt(0);
#endif
  }
#else
  {
    const uint4* src = (const uint4*)(ws + WS_W12);
    uint4* dst = (uint4*)(smem + SM_WFRAG);
    for (int i = tid; i < (160 * 1024) / 16; i += WAVES * 32) dst[i] = src[i];
  }
#endif
  __syncthreads();

  // ---- per-wave scratch ----
  char*  wbase = smem + SM_WAVE0 + wv * PW;
  f16*   embw  = (f16*)(wbase + O_EMB);     // [16][136]
  f16*   vnw   = (f16*)(wbase + O_EMB);     // alias: [64][64]
  float* kbuf  = (float*)(wbase + O_KH);    // [16][66]
  f16*   Hs    = (f16*)(wbase + O_KH);      // alias: [16][36]
  float* gatep = (float*)(wbase + O_GATE);  // [16]
  f16*   valw  = (f16*)(wbase + O_VAL);     // [64][64]
  const f16* xt = (const f16*)(smem + SM_XT);

  const int t0 = g0 + wv * WOUT;
  const int Mr = (lane & 16) ? 8 : 0;       // C/D row base for this lane half
  const v8f z8f = { 0.f, 0.f, 0.f, 0.f, 0.f, 0.f, 0.f, 0.f };

  // ---- 4 M-tiles: 1 halo tile (t0-16..t0-1) + 3 output tiles ----
  for (int tile = 0; tile < 4; ++tile) {
    const int trow0 = t0 - 16 + tile * 16;
    const int xr0   = trow0 - g0 + 18;      // x-tile row of token trow0

    // warm L0/WGP$: this tile's hidden rows for the gate (2 lines per row)
    {
      int row = lane & 15, t = trow0 + row;
      int tc = t < 0 ? 0 : (t >= TLEN ? TLEN - 1 : t);
      __builtin_prefetch(hidden + ((size_t)b * TLEN + tc) * D + (lane >> 4) * 32, 0, 3);
    }
    // one-shot warm of the streamed key/value fragment arrays (16KB each)
    if (tile == 0) {
#pragma unroll
      for (int q = 0; q < 4; ++q) {
        __builtin_prefetch(ws + WS_KEY + lane * 512 + q * 128, 0, 3);
        __builtin_prefetch(ws + WS_VAL + lane * 512 + q * 128, 0, 3);
      }
    }

    // ====== engram MLPs (m=0: n=2, m=1: n=3), fused GEMM1->silu->GEMM2 ======
    for (int m = 0; m < 2; ++m) {
      const int   KT  = m ? 6 : 4;
      const char* w1f = (const char*)smem + (m ? SM_W13 : SM_WFRAG);
      const char* w2f = ws + (m ? WS_W23 : WS_W22);
      const float* b1p = m ? b13 : b12;
      const float* b2p = m ? b23 : b22;
      const f16* xrow = xt + (xr0 - (m + 1)) * 64;   // window base row: t-(n-1)
      v8f acc[4];
#pragma unroll
      for (int nt = 0; nt < 4; ++nt) acc[nt] = z8f;

      // warm first K-chunk of the streamed W2 fragments
#pragma unroll
      for (int nt = 0; nt < 4; ++nt)
        __builtin_prefetch(w2f + (nt * 8) * 1024 + lane * 32, 0, 3);

      for (int jn = 0; jn < 16; jn += 2) {           // hidden cols 32 at a time
        int kc = jn >> 1;
        // prefetch next K-chunk's W2 fragments one iteration ahead
        if (kc < 7) {
#pragma unroll
          for (int nt = 0; nt < 4; ++nt)
            __builtin_prefetch(w2f + (nt * 8 + kc + 1) * 1024 + lane * 32, 0, 3);
        }
        v8f h0 = z8f, h1 = z8f;
        for (int kt = 0; kt < KT; ++kt) {
          v16h a  = load_afrag_win(xrow, kt * 32, lane);
          v16h bb0 = load_bfrag(w1f + ((jn    ) * KT + kt) * 1024, lane);
          v16h bb1 = load_bfrag(w1f + ((jn + 1) * KT + kt) * 1024, lane);
          h0 = wmma16(a, bb0, h0);
          h1 = wmma16(a, bb1, h1);
        }
        float bs0 = b1p[jn * 16 + (lane & 15)];
        float bs1 = b1p[jn * 16 + 16 + (lane & 15)];
#pragma unroll
        for (int r = 0; r < 8; ++r) {                // bias + SiLU -> K-panel
          Hs[(r + Mr) * 36 + (lane & 15)]      = (f16)siluf(h0[r] + bs0);
          Hs[(r + Mr) * 36 + 16 + (lane & 15)] = (f16)siluf(h1[r] + bs1);
        }
        v16h a2 = load_afrag_row(Hs, 36, lane);      // immediately consume
#pragma unroll
        for (int nt = 0; nt < 4; ++nt) {
          v16h bb = load_bfrag(w2f + (nt * 8 + kc) * 1024, lane);
          acc[nt] = wmma16(a2, bb, acc[nt]);
        }
      }
#pragma unroll
      for (int nt = 0; nt < 4; ++nt) {               // emb = [out2 | out3]
        float b2s = b2p[nt * 16 + (lane & 15)];
        int col = m * 64 + nt * 16 + (lane & 15);
#pragma unroll
        for (int r = 0; r < 8; ++r)
          embw[(r + Mr) * 136 + col] = (f16)(acc[nt][r] + b2s);
      }
    }

    // ====== key projection ======
    {
      v8f ka[4];
#pragma unroll
      for (int nt = 0; nt < 4; ++nt) ka[nt] = z8f;
      for (int kt = 0; kt < 4; ++kt) {
        v16h a = load_afrag_row(embw + kt * 32, 136, lane);
#pragma unroll
        for (int nt = 0; nt < 4; ++nt) {
          v16h bb = load_bfrag(ws + WS_KEY + (nt * 4 + kt) * 1024, lane);
          ka[nt] = wmma16(a, bb, ka[nt]);
        }
      }
#pragma unroll
      for (int nt = 0; nt < 4; ++nt) {
        float kb2 = keyb[nt * 16 + (lane & 15)];
#pragma unroll
        for (int r = 0; r < 8; ++r)
          kbuf[(r + Mr) * 66 + nt * 16 + (lane & 15)] = ka[nt][r] + kb2;
      }
    }

    // ====== gate: sigmoid(signsqrt(dot(LN(k), LN(q)) / 8)) ======
    if (lane < 16) {
      int row = lane, t = trow0 + row;
      int tc = t < 0 ? 0 : (t >= TLEN ? TLEN - 1 : t);
      const float* hp = hidden + ((size_t)b * TLEN + tc) * D;
      const float* kr = kbuf + row * 66;
      float sk = 0.f, sk2 = 0.f, sh = 0.f, sh2 = 0.f;
      for (int i = 0; i < D; ++i) {
        float kv = kr[i], hv = hp[i];
        sk += kv; sk2 += kv * kv; sh += hv; sh2 += hv * hv;
      }
      float mk = sk * (1.f / D), mh = sh * (1.f / D);
      float rk = rsqrtf(sk2 * (1.f / D) - mk * mk + 1e-5f);
      float rh = rsqrtf(sh2 * (1.f / D) - mh * mh + 1e-5f);
      float dot = 0.f;
      for (int i = 0; i < D; ++i) {
        float nk = (kr[i] - mk) * rk * n1g[i] + n1b[i];
        float nq = (hp[i] - mh) * rh * n2g[i] + n2b[i];
        dot += nk * nq;
      }
      float g = dot * 0.125f;
      float sq = copysignf(sqrtf(fmaxf(fabsf(g), 1e-6f)), g);
      gatep[row] = 1.f / (1.f + __expf(-sq));
    }

    // ====== value projection * gate ======
    {
      v8f va[4];
#pragma unroll
      for (int nt = 0; nt < 4; ++nt) va[nt] = z8f;
      for (int kt = 0; kt < 4; ++kt) {
        v16h a = load_afrag_row(embw + kt * 32, 136, lane);
#pragma unroll
        for (int nt = 0; nt < 4; ++nt) {
          v16h bb = load_bfrag(ws + WS_VAL + (nt * 4 + kt) * 1024, lane);
          va[nt] = wmma16(a, bb, va[nt]);
        }
      }
#pragma unroll
      for (int nt = 0; nt < 4; ++nt) {
        float vb2 = valb[nt * 16 + (lane & 15)];
#pragma unroll
        for (int r = 0; r < 8; ++r) {
          int M = r + Mr, t = trow0 + M;
          float v = (va[nt][r] + vb2) * gatep[M];
          if (t < 0) v = 0.f;                        // causal zero padding
          valw[(tile * 16 + M) * 64 + nt * 16 + (lane & 15)] = (f16)v;
        }
      }
    }
  }

  // ====== conv_norm(value) for all 64 rows (halo included) ======
#pragma unroll
  for (int p = 0; p < 2; ++p) {
    int vr = lane + p * 32;
    int t = t0 - 16 + vr;
    const f16* vrw = valw + vr * 64;
    float s = 0.f, s2 = 0.f;
    for (int i = 0; i < D; ++i) { float v = (float)vrw[i]; s += v; s2 += v * v; }
    float mm = s * (1.f / D);
    float rs = rsqrtf(s2 * (1.f / D) - mm * mm + 1e-5f);
    for (int i = 0; i < D; ++i) {
      float v = ((float)vrw[i] - mm) * rs * cng[i] + cnb[i];
      vnw[vr * 64 + i] = (t < 0) ? (f16)0.f : (f16)v;   // conv zero-pads t<0
    }
  }

  // ====== dilated depthwise conv (K=4, dil=3) + SiLU + residual ======
  {
    int d0 = lane, d1 = lane + 32;
    float wa0 = convw[d0*4+0], wa1 = convw[d0*4+1], wa2 = convw[d0*4+2], wa3 = convw[d0*4+3];
    float wb0 = convw[d1*4+0], wb1 = convw[d1*4+1], wb2 = convw[d1*4+2], wb3 = convw[d1*4+3];
    for (int o = 0; o < WOUT; ++o) {
      int t = t0 + o;
      if (t >= TLEN) break;
      int vr = 16 + o;
      float y0 = wa0 * (float)vnw[(vr-9)*64+d0] + wa1 * (float)vnw[(vr-6)*64+d0]
               + wa2 * (float)vnw[(vr-3)*64+d0] + wa3 * (float)vnw[vr*64+d0];
      float y1 = wb0 * (float)vnw[(vr-9)*64+d1] + wb1 * (float)vnw[(vr-6)*64+d1]
               + wb2 * (float)vnw[(vr-3)*64+d1] + wb3 * (float)vnw[vr*64+d1];
      size_t ob = ((size_t)b * TLEN + t) * D;
      out[ob + d0] = (float)valw[vr*64+d0] + siluf(y0);
      out[ob + d1] = (float)valw[vr*64+d1] + siluf(y1);
    }
  }
}

// ================= host =====================================================
extern "C" void kernel_launch(void* const* d_in, const int* in_sizes, int n_in,
                              void* d_out, int out_size, void* d_ws, size_t ws_size,
                              hipStream_t stream) {
  const float* hidden = (const float*)d_in[0];
  const int*   ids    = (const int*)d_in[1];
  const float* emb    = (const float*)d_in[2];
  const float* W12 = (const float*)d_in[3],  *b12 = (const float*)d_in[4];
  const float* W22 = (const float*)d_in[5],  *b22 = (const float*)d_in[6];
  const float* W13 = (const float*)d_in[7],  *b13 = (const float*)d_in[8];
  const float* W23 = (const float*)d_in[9],  *b23 = (const float*)d_in[10];
  const float* keyW = (const float*)d_in[11], *keyb = (const float*)d_in[12];
  const float* valW = (const float*)d_in[13], *valb = (const float*)d_in[14];
  const float* n1g = (const float*)d_in[15], *n1b = (const float*)d_in[16];
  const float* n2g = (const float*)d_in[17], *n2b = (const float*)d_in[18];
  const float* convw = (const float*)d_in[19];
  const float* cng = (const float*)d_in[20], *cnb = (const float*)d_in[21];
  char* ws = (char*)d_ws;

  // prep: 256 fragment blocks + 63 blocks for the f16 embedding table
  engram_prep<<<dim3(319), 256, 0, stream>>>(W12, W13, W22, W23, keyW, valW, emb, ws);

  hipFuncSetAttribute((const void*)engram_kernel,
                      hipFuncAttributeMaxDynamicSharedMemorySize, SMEM_TOTAL);
  dim3 grid((TLEN + WG_TOK - 1) / WG_TOK, BATCH);
  engram_kernel<<<grid, WAVES * 32, SMEM_TOTAL, stream>>>(
      hidden, ids, ws, b12, b22, b13, b23, keyb, valb,
      n1g, n1b, n2g, n2b, convw, cng, cnb, (float*)d_out);
}
